// ConvAttention_23527830847666
// MI455X (gfx1250) — matrix operand order repaired and checked by reading, not compiled, not run on hardware
//
#include <hip/hip_runtime.h>

typedef __attribute__((ext_vector_type(16))) _Float16 v16h;
typedef __attribute__((ext_vector_type(8)))  _Float16 v8h;
typedef __attribute__((ext_vector_type(8)))  float    v8f;

#define DIM     512
#define NPOS    1024      // 32*32
#define BATCH   8
#define HEADS   8
#define DHEAD   64
#define QKV_M   1536
#define LSTR    72        // 64 + 8 pad, multiple of 8 halves (16B aligned rows)
#define PSTR    136       // 128 + 8 pad

// ---- CDNA5 async global->LDS staging (ASYNCcnt-tracked DMA path) -------
// GLOBAL_LOAD_ASYNC_TO_LDS_B128: vdst = LDS byte offset, vaddr = 64-bit
// global address (GV mode). LDS offset == low 32 bits of generic shared ptr
__device__ inline void async_copy16(_Float16* lds_dst, const _Float16* g_src) {
  unsigned lo = (unsigned)(size_t)lds_dst;
  unsigned long long ga = (unsigned long long)g_src;
  asm volatile("global_load_async_to_lds_b128 %0, %1, off"
               :: "v"(lo), "v"(ga) : "memory");
}
__device__ inline void async_wait() {
  asm volatile("s_wait_asynccnt 0x0" ::: "memory");
}

// ---- WMMA helpers ------------------------------------------------------
__device__ inline v8f wmma_f16(v16h a, v16h b, v8f c) {
  return __builtin_amdgcn_wmma_f32_16x16x32_f16(false, a, false, b, (short)0, c, false, false);
}

// A-matrix 16x32 f16 fragment (ISA 7.12.2): lanes 0-15 -> M=lane, K 0-7 in v0..3,
// K 16-23 in v4..7; lanes 16-31 -> K 8-15 / 24-31. Row-major K-contiguous LDS.
__device__ inline v16h frag_a(const _Float16* base, int stride, int koff) {
  int lane = threadIdx.x & 31;
  const _Float16* p = base + (lane & 15) * stride + koff + ((lane >> 4) << 3);
  v8h lo = *(const v8h*)p;         // K = kb .. kb+7
  v8h hi = *(const v8h*)(p + 16);  // K = kb+16 .. kb+23
  return __builtin_shufflevector(lo, hi, 0,1,2,3,4,5,6,7,8,9,10,11,12,13,14,15);
}

// B-matrix 32x16 f16 fragment: lane holds column N=lane&15; lanes 0-15 K 0-7 /
// 16-23, lanes 16-31 K 8-15 / 24-31. LDS layout is B^T: [n][k], K-contiguous.
__device__ inline v16h frag_b(const _Float16* base, int stride, int koff) {
  int lane = threadIdx.x & 31;
  const _Float16* p = base + (lane & 15) * stride + koff + ((lane >> 4) << 3);
  v8h lo = *(const v8h*)p;
  v8h hi = *(const v8h*)(p + 16);
  return __builtin_shufflevector(lo, hi, 0,1,2,3,4,5,6,7,8,9,10,11,12,13,14,15);
}

// ---- K1: channel LayerNorm -> f16 activations, layout [b][n][c] --------
__global__ __launch_bounds__(256) void ln_kernel(const float* __restrict__ x,
                                                 const float* __restrict__ gamma,
                                                 _Float16* __restrict__ Xn) {
  int idx = blockIdx.x * 256 + threadIdx.x;        // 0..8191 = b*1024 + pos
  int b = idx >> 10, p = idx & 1023;
  const float* xb = x + ((size_t)b * DIM) * NPOS + p;
  float s = 0.f, s2 = 0.f;
  for (int c = 0; c < DIM; ++c) {
    float v = xb[(size_t)c * NPOS];
    s += v; s2 += v * v;
  }
  float mean = s * (1.f / DIM);
  float var  = s2 * (1.f / DIM) - mean * mean;     // biased, as jnp.var
  float rstd = rsqrtf(var + 1e-5f);
  _Float16* o = Xn + ((size_t)b * NPOS + p) * DIM;
  for (int c = 0; c < DIM; ++c)
    o[c] = (_Float16)((xb[(size_t)c * NPOS] - mean) * rstd * gamma[c]);
}

// ---- K2: QKV GEMM  C[o][n] = w_qkv(1536x512) @ Xn^T ---------------------
// Q,K stored [b][n][c] (K-major for attention A/B frags); V stored [b][c][n].
__global__ __launch_bounds__(256) void qkv_gemm(const float* __restrict__ W,
                                                const _Float16* __restrict__ Xn,
                                                _Float16* __restrict__ Qb,
                                                _Float16* __restrict__ Kb,
                                                _Float16* __restrict__ Vb) {
  __shared__ __align__(16) _Float16 As[128 * LSTR];
  __shared__ __align__(16) _Float16 Bs[128 * LSTR];
  int b  = blockIdx.z;
  int m0 = blockIdx.x * 128;
  int n0 = blockIdx.y * 128;
  int tid = threadIdx.x, wave = tid >> 5;
  int wm = wave & 3, wn = wave >> 2;               // waves: 4 (M) x 2 (N)
  const _Float16* xb = Xn + (size_t)b * NPOS * DIM;
  v8f acc[2][4] = {};

  for (int k0 = 0; k0 < DIM; k0 += 64) {
    for (int i = tid; i < 128 * 8; i += 256) {     // A: f32 -> f16 stage (VALU)
      int r = i >> 3, kc = (i & 7) * 8;
      const float* src = W + (size_t)(m0 + r) * DIM + k0 + kc;
      v8h h;
      #pragma unroll
      for (int e = 0; e < 8; ++e) h[e] = (_Float16)src[e];
      *(v8h*)&As[r * LSTR + kc] = h;
    }
    for (int i = tid; i < 128 * 8; i += 256) {     // B stage: async DMA to LDS
      int r = i >> 3, kc = (i & 7) * 8;
      async_copy16(&Bs[r * LSTR + kc], xb + (size_t)(n0 + r) * DIM + k0 + kc);
    }
    if (k0 + 64 < DIM)                             // global_prefetch_b8 next A tile
      __builtin_prefetch(W + (size_t)(m0 + (tid >> 1)) * DIM + k0 + 64 + (tid & 1) * 32, 0, 0);
    async_wait();
    __syncthreads();
    #pragma unroll
    for (int kk = 0; kk < 64; kk += 32) {
      v16h af[2], bf[4];
      #pragma unroll
      for (int i = 0; i < 2; ++i) af[i] = frag_a(&As[(wm * 32 + i * 16) * LSTR], LSTR, kk);
      #pragma unroll
      for (int j = 0; j < 4; ++j) bf[j] = frag_b(&Bs[(wn * 64 + j * 16) * LSTR], LSTR, kk);
      #pragma unroll
      for (int i = 0; i < 2; ++i)
        #pragma unroll
        for (int j = 0; j < 4; ++j)
          acc[i][j] = wmma_f16(af[i], bf[j], acc[i][j]);
    }
    __syncthreads();
  }

  int lane = tid & 31, col = lane & 15, rhi = (lane >> 4) << 3;
  #pragma unroll
  for (int i = 0; i < 2; ++i)
    #pragma unroll
    for (int j = 0; j < 4; ++j)
      #pragma unroll
      for (int r = 0; r < 8; ++r) {
        int gm = m0 + wm * 32 + i * 16 + rhi + r;
        int gn = n0 + wn * 64 + j * 16 + col;
        _Float16 hv = (_Float16)acc[i][j][r];
        if (gm < DIM)
          Qb[((size_t)b * NPOS + gn) * DIM + gm] = hv;
        else if (gm < 2 * DIM)
          Kb[((size_t)b * NPOS + gn) * DIM + (gm - DIM)] = hv;
        else
          Vb[((size_t)b * DIM + (gm - 2 * DIM)) * NPOS + gn] = hv;
      }
}

// ---- K3: flash attention per (b, h, 128-row i-strip) --------------------
__global__ __launch_bounds__(256) void attn_kernel(const _Float16* __restrict__ Qb,
                                                   const _Float16* __restrict__ Kb,
                                                   const _Float16* __restrict__ Vb,
                                                   _Float16* __restrict__ Ob) {
  __shared__ __align__(16) _Float16 Qs[128 * LSTR];
  __shared__ __align__(16) _Float16 Ks[128 * LSTR];
  __shared__ __align__(16) _Float16 Vs[DHEAD * PSTR];
  __shared__ __align__(16) _Float16 Ps[8 * 16 * PSTR];   // per-wave P strip
  int b = blockIdx.z, h = blockIdx.y;
  int i0 = blockIdx.x * 128;
  int tid = threadIdx.x, wave = tid >> 5, lane = tid & 31;
  int col = lane & 15, hi = lane >> 4;
  const size_t qkbase = (size_t)b * NPOS * DIM + (size_t)h * DHEAD;
  const size_t vbase  = ((size_t)b * DIM + (size_t)h * DHEAD) * NPOS;

  for (int i = tid; i < 128 * 8; i += 256) {             // stage Q (async DMA)
    int r = i >> 3, kc = (i & 7) * 8;
    async_copy16(&Qs[r * LSTR + kc], Qb + qkbase + (size_t)(i0 + r) * DIM + kc);
  }

  float mstate[8], lstate[8];
  #pragma unroll
  for (int r = 0; r < 8; ++r) { mstate[r] = -1e30f; lstate[r] = 0.f; }
  v8f oacc[4] = {};
  _Float16* Pw = &Ps[wave * 16 * PSTR];

  for (int j0 = 0; j0 < NPOS; j0 += 128) {
    __syncthreads();                                     // Ks/Vs reuse fence
    for (int i = tid; i < 128 * 8; i += 256) {           // K tile [j][c] async
      int r = i >> 3, kc = (i & 7) * 8;
      async_copy16(&Ks[r * LSTR + kc], Kb + qkbase + (size_t)(j0 + r) * DIM + kc);
    }
    for (int i = tid; i < DHEAD * 16; i += 256) {        // V tile [c][j] async
      int c = i >> 4, jc = (i & 15) * 8;
      async_copy16(&Vs[c * PSTR + jc], Vb + vbase + (size_t)c * NPOS + j0 + jc);
    }
    async_wait();                                        // drains K/V (and Q once)
    __syncthreads();

    // sim strip: wave's 16 q-rows x 128 k-cols, K=64 (2 wmma per subtile)
    v16h qf0 = frag_a(&Qs[(wave * 16) * LSTR], LSTR, 0);
    v16h qf1 = frag_a(&Qs[(wave * 16) * LSTR], LSTR, 32);
    v8f s[8] = {};
    #pragma unroll
    for (int nt = 0; nt < 8; ++nt) {
      v16h kf0 = frag_b(&Ks[(nt * 16) * LSTR], LSTR, 0);
      v16h kf1 = frag_b(&Ks[(nt * 16) * LSTR], LSTR, 32);
      s[nt] = wmma_f16(qf0, kf0, s[nt]);
      s[nt] = wmma_f16(qf1, kf1, s[nt]);
    }

    // online softmax; D-layout row = r + 8*hi lives across one 16-lane half
    float corr[8];
    #pragma unroll
    for (int r = 0; r < 8; ++r) {
      float mx = -1e30f;
      #pragma unroll
      for (int nt = 0; nt < 8; ++nt) mx = fmaxf(mx, s[nt][r] * 0.125f);
      #pragma unroll
      for (int m = 8; m >= 1; m >>= 1) mx = fmaxf(mx, __shfl_xor(mx, m, 32));
      float mnew = fmaxf(mstate[r], mx);
      corr[r] = __expf(mstate[r] - mnew);
      float rs = 0.f;
      #pragma unroll
      for (int nt = 0; nt < 8; ++nt) {
        float p = __expf(s[nt][r] * 0.125f - mnew);
        s[nt][r] = p;
        rs += p;
      }
      #pragma unroll
      for (int m = 8; m >= 1; m >>= 1) rs += __shfl_xor(rs, m, 32);
      lstate[r] = lstate[r] * corr[r] + rs;
      mstate[r] = mnew;
    }

    // D-layout -> A-layout transpose of P through per-wave LDS (in-order DS)
    #pragma unroll
    for (int nt = 0; nt < 8; ++nt)
      #pragma unroll
      for (int r = 0; r < 8; ++r)
        Pw[(hi * 8 + r) * PSTR + nt * 16 + col] = (_Float16)s[nt][r];

    #pragma unroll
    for (int ct = 0; ct < 4; ++ct)
      #pragma unroll
      for (int r = 0; r < 8; ++r)
        oacc[ct][r] *= corr[r];

    #pragma unroll
    for (int kt = 0; kt < 4; ++kt) {                     // out += P @ V^T
      v16h pf = frag_a(Pw, PSTR, kt * 32);
      #pragma unroll
      for (int ct = 0; ct < 4; ++ct) {
        v16h vf = frag_b(&Vs[(ct * 16) * PSTR], PSTR, kt * 32);
        oacc[ct] = wmma_f16(pf, vf, oacc[ct]);
      }
    }
  }

  size_t obase = (size_t)b * NPOS * DIM + (size_t)h * DHEAD;
  #pragma unroll
  for (int ct = 0; ct < 4; ++ct)
    #pragma unroll
    for (int r = 0; r < 8; ++r) {
      int row = i0 + wave * 16 + hi * 8 + r;
      Ob[obase + (size_t)row * DIM + ct * 16 + col] =
          (_Float16)(oacc[ct][r] / lstate[r]);
    }
}

// ---- K4: output projection + residual ----------------------------------
__global__ __launch_bounds__(256) void out_gemm(const float* __restrict__ W,
                                                const _Float16* __restrict__ Ob,
                                                const float* __restrict__ x,
                                                float* __restrict__ out) {
  __shared__ __align__(16) _Float16 As[128 * LSTR];
  __shared__ __align__(16) _Float16 Bs[128 * LSTR];
  int b  = blockIdx.z;
  int m0 = blockIdx.x * 128;
  int n0 = blockIdx.y * 128;
  int tid = threadIdx.x, wave = tid >> 5;
  int wm = wave & 3, wn = wave >> 2;
  const _Float16* ob = Ob + (size_t)b * NPOS * DIM;
  v8f acc[2][4] = {};

  for (int k0 = 0; k0 < DIM; k0 += 64) {
    for (int i = tid; i < 128 * 8; i += 256) {
      int r = i >> 3, kc = (i & 7) * 8;
      const float* src = W + (size_t)(m0 + r) * DIM + k0 + kc;
      v8h hv;
      #pragma unroll
      for (int e = 0; e < 8; ++e) hv[e] = (_Float16)src[e];
      *(v8h*)&As[r * LSTR + kc] = hv;
    }
    for (int i = tid; i < 128 * 8; i += 256) {     // B stage: async DMA to LDS
      int r = i >> 3, kc = (i & 7) * 8;
      async_copy16(&Bs[r * LSTR + kc], ob + (size_t)(n0 + r) * DIM + k0 + kc);
    }
    if (k0 + 64 < DIM)
      __builtin_prefetch(W + (size_t)(m0 + (tid >> 1)) * DIM + k0 + 64 + (tid & 1) * 32, 0, 0);
    async_wait();
    __syncthreads();
    #pragma unroll
    for (int kk = 0; kk < 64; kk += 32) {
      v16h af[2], bf[4];
      #pragma unroll
      for (int i = 0; i < 2; ++i) af[i] = frag_a(&As[(wm * 32 + i * 16) * LSTR], LSTR, kk);
      #pragma unroll
      for (int j = 0; j < 4; ++j) bf[j] = frag_b(&Bs[(wn * 64 + j * 16) * LSTR], LSTR, kk);
      #pragma unroll
      for (int i = 0; i < 2; ++i)
        #pragma unroll
        for (int j = 0; j < 4; ++j)
          acc[i][j] = wmma_f16(af[i], bf[j], acc[i][j]);
    }
    __syncthreads();
  }

  int lane = tid & 31, col = lane & 15, rhi = (lane >> 4) << 3;
  #pragma unroll
  for (int i = 0; i < 2; ++i)
    #pragma unroll
    for (int j = 0; j < 4; ++j)
      #pragma unroll
      for (int r = 0; r < 8; ++r) {
        int gm = m0 + wm * 32 + i * 16 + rhi + r;
        int gn = n0 + wn * 64 + j * 16 + col;
        size_t idx = ((size_t)b * DIM + gm) * NPOS + gn;
        out[idx] = acc[i][j][r] + x[idx];
      }
}

extern "C" void kernel_launch(void* const* d_in, const int* in_sizes, int n_in,
                              void* d_out, int out_size, void* d_ws, size_t ws_size,
                              hipStream_t stream) {
  const float* x     = (const float*)d_in[0];
  const float* gamma = (const float*)d_in[1];
  const float* w_qkv = (const float*)d_in[2];
  const float* w_out = (const float*)d_in[3];
  float* out = (float*)d_out;

  char* ws = (char*)d_ws;
  const size_t SZ = (size_t)BATCH * NPOS * DIM * 2;   // 8 MB per f16 tensor
  _Float16* Xn = (_Float16*)(ws);
  _Float16* Qb = (_Float16*)(ws + SZ);
  _Float16* Kb = (_Float16*)(ws + 2 * SZ);
  _Float16* Vb = (_Float16*)(ws + 3 * SZ);
  _Float16* Ob = (_Float16*)(ws + 4 * SZ);

  ln_kernel<<<dim3((BATCH * NPOS) / 256), 256, 0, stream>>>(x, gamma, Xn);
  qkv_gemm<<<dim3(QKV_M / 128, NPOS / 128, BATCH), 256, 0, stream>>>(w_qkv, Xn, Qb, Kb, Vb);
  attn_kernel<<<dim3(NPOS / 128, HEADS, BATCH), 256, 0, stream>>>(Qb, Kb, Vb, Ob);
  out_gemm<<<dim3(DIM / 128, NPOS / 128, BATCH), 256, 0, stream>>>(w_out, Ob, x, out);
}
